// VLM_Updater_25426206392332
// MI455X (gfx1250) — compile-verified, matrix operand order checked by reading, not run on hardware
//
#include <hip/hip_runtime.h>

typedef __attribute__((ext_vector_type(16))) _Float16 v16h;
typedef __attribute__((ext_vector_type(8)))  float    v8f;

#define DD 1536
#define EE 1024
#define NB 8
#define NN 2048
#define KTILES 48      // 1536 / 32
#define NTILES 96      // 1536 / 16
#define ASTR 1544      // halves per A row in LDS (stride 772 dwords, %64==4 -> no bank clash)
#define PROJ_BLOCKS (NB * 6)                       // 48
#define WCONV_BLOCKS ((NB * KTILES * NTILES * 32) / 256)   // 4608

// ---------------------------------------------------------------------------
// Kernel 1 (fused grid): blocks [0,48): img projection p = img@Wp + bp
//                        blocks [48,...): gather/convert/shuffle Wf_top -> f16
// W16s layout: [b][kt][nt][lane][16 halves]  (B-fragment-major, 32B/lane)
// B fragment (32x16 f16): lane -> n = nt*16+(lane&15), h=lane>>4;
// element i=2v+p -> k = kt*32 + (v<4 ? 2v+8h : 16+2(v-4)+8h) + p.
// ---------------------------------------------------------------------------
__global__ __launch_bounds__(256) void k_prep(
    const float* __restrict__ img, const float* __restrict__ Wp,
    const float* __restrict__ bp,
    const float* __restrict__ Wf, const int* __restrict__ cat_ids,
    float* __restrict__ p_vec, _Float16* __restrict__ W16s) {
  __shared__ float s_img[EE];
  const int t = threadIdx.x;

  if (blockIdx.x < PROJ_BLOCKS) {
    // ---- projection: one block per (b, 256-col segment) ----
    const int b = blockIdx.x / 6, seg = blockIdx.x % 6;
    const int d = seg * 256 + t;
    for (int i = t; i < EE; i += 256) s_img[i] = img[b * EE + i];
    __syncthreads();
    float acc = bp[d];
    #pragma unroll 4
    for (int e = 0; e < EE; ++e) acc = fmaf(s_img[e], Wp[(size_t)e * DD + d], acc);
    p_vec[b * DD + d] = acc;
    return;
  }

  // ---- weight gather + f32->f16 + fragment shuffle ----
  const long e = (long)(blockIdx.x - PROJ_BLOCKS) * 256 + t;  // < 1,179,648
  const int lane = (int)(e & 31);
  long f = e >> 5;
  const int nt = (int)(f % NTILES); f /= NTILES;
  const int kt = (int)(f % KTILES); f /= KTILES;
  const int b  = (int)f;
  const int cat = cat_ids[b];
  const int n = nt * 16 + (lane & 15);
  const int h = lane >> 4;
  const float* __restrict__ W = Wf + (size_t)cat * (2 * DD) * DD;  // top half

  v16h frag;
  #pragma unroll
  for (int v = 0; v < 8; ++v) {
    const int kb = (v < 4) ? (2 * v + 8 * h) : (16 + 2 * (v - 4) + 8 * h);
    const int k = kt * 32 + kb;
    frag[2 * v]     = (_Float16)W[(size_t)k * DD + n];
    frag[2 * v + 1] = (_Float16)W[(size_t)(k + 1) * DD + n];
  }
  *(v16h*)(W16s + (size_t)e * 16) = frag;
}

// ---------------------------------------------------------------------------
// Kernel 2: a[b,:] = LN_attn( clip(LN_img(p), +-10) * 100 ).  8 blocks.
// ---------------------------------------------------------------------------
__global__ __launch_bounds__(256) void k_ln(
    const float* __restrict__ p_vec,
    const float* __restrict__ g_img, const float* __restrict__ b_img,
    const float* __restrict__ g_attn, const float* __restrict__ b_attn,
    float* __restrict__ a_vec) {
  const int b = blockIdx.x, t = threadIdx.x;
  __shared__ float r1[256], r2[256];

  float p[6];
  #pragma unroll
  for (int j = 0; j < 6; ++j) p[j] = p_vec[b * DD + t + 256 * j];

  float s = 0.f, s2 = 0.f;
  #pragma unroll
  for (int j = 0; j < 6; ++j) { s += p[j]; s2 += p[j] * p[j]; }
  r1[t] = s; r2[t] = s2; __syncthreads();
  for (int off = 128; off >= 1; off >>= 1) {
    if (t < off) { r1[t] += r1[t + off]; r2[t] += r2[t + off]; }
    __syncthreads();
  }
  float mean = r1[0] * (1.f / DD);
  float var  = r2[0] * (1.f / DD) - mean * mean;
  float rstd = rsqrtf(var + 1e-5f);
  __syncthreads();

  float y[6];
  #pragma unroll
  for (int j = 0; j < 6; ++j) {
    const int d = t + 256 * j;
    float x = (p[j] - mean) * rstd * g_img[d] + b_img[d];
    x = fminf(fmaxf(x, -10.f), 10.f) * 100.f;
    x = fminf(fmaxf(x, -1000.f), 1000.f);
    y[j] = x;
  }

  s = 0.f; s2 = 0.f;
  #pragma unroll
  for (int j = 0; j < 6; ++j) { s += y[j]; s2 += y[j] * y[j]; }
  r1[t] = s; r2[t] = s2; __syncthreads();
  for (int off = 128; off >= 1; off >>= 1) {
    if (t < off) { r1[t] += r1[t + off]; r2[t] += r2[t + off]; }
    __syncthreads();
  }
  mean = r1[0] * (1.f / DD);
  var  = r2[0] * (1.f / DD) - mean * mean;
  rstd = rsqrtf(var + 1e-5f);

  #pragma unroll
  for (int j = 0; j < 6; ++j) {
    const int d = t + 256 * j;
    a_vec[b * DD + d] = (y[j] - mean) * rstd * g_attn[d] + b_attn[d];
  }
}

// ---------------------------------------------------------------------------
// Kernel 3: c[b,:] = a[b,:] @ Wf_bot[cat_b] + bf[cat_b]   (f32)
// ---------------------------------------------------------------------------
__global__ __launch_bounds__(256) void k_cvec(
    const float* __restrict__ a_vec, const float* __restrict__ Wf,
    const float* __restrict__ bf, const int* __restrict__ cat_ids,
    float* __restrict__ c_vec) {
  const int b = blockIdx.x / 6, seg = blockIdx.x % 6, t = threadIdx.x;
  const int d = seg * 256 + t;
  __shared__ float s_a[DD];
  for (int i = t; i < DD; i += 256) s_a[i] = a_vec[b * DD + i];
  __syncthreads();
  const int cat = cat_ids[b];
  const float* __restrict__ Wb = Wf + ((size_t)cat * (2 * DD) + DD) * DD;
  float acc = bf[cat * DD + d];
  #pragma unroll 4
  for (int k = 0; k < DD; ++k) acc = fmaf(s_a[k], Wb[(size_t)k * DD + d], acc);
  c_vec[b * DD + d] = acc;
}

// ---------------------------------------------------------------------------
// Kernel 4: fused = pre_vlm @ Wf_top + c, then LayerNorm, register-resident.
// Grid 512 M-tiles of 32 rows. 384 threads = 12 waves.
// Wave w: N-tiles nt = w*8..w*8+7, both 16-row M-subtiles (register B reuse:
// each B fragment feeds 2 WMMAs). acc = 2x8 v8f = 128 VGPRs.
// ---------------------------------------------------------------------------
__global__ __launch_bounds__(384) void k_gemm(
    const float* __restrict__ pre_vlm, const _Float16* __restrict__ W16s,
    const float* __restrict__ c_vec,
    const float* __restrict__ g_fuse, const float* __restrict__ b_fuse,
    float* __restrict__ out) {
  __shared__ __align__(16) _Float16 As[32 * ASTR];   // 98,816 B
  __shared__ float red1[12 * 32], red2[12 * 32];
  __shared__ float tot1[32], tot2[32];

  const int mt = blockIdx.x;            // 0..511, 64 tiles/batch
  const int b  = mt >> 6;
  const int t = threadIdx.x, lane = t & 31, w = t >> 5;
  const int m0 = mt * 32;

  // ---- stage A tile (32 x 1536) f32 -> f16 in LDS ----
  const float* __restrict__ Ag = pre_vlm + (size_t)m0 * DD;
  for (int i = t; i < 32 * (DD / 4); i += 384) {
    const int r = i / (DD / 4), c4 = i % (DD / 4);
    const float4 v = ((const float4*)(Ag + (size_t)r * DD))[c4];
    _Float16* dst = As + r * ASTR + c4 * 4;
    dst[0] = (_Float16)v.x; dst[1] = (_Float16)v.y;
    dst[2] = (_Float16)v.z; dst[3] = (_Float16)v.w;
  }
  __syncthreads();

  v8f acc[2][8];
  const v8f vzero = {0.f, 0.f, 0.f, 0.f, 0.f, 0.f, 0.f, 0.f};
  #pragma unroll
  for (int mg = 0; mg < 2; ++mg)
    #pragma unroll
    for (int i = 0; i < 8; ++i) acc[mg][i] = vzero;

  const int mrow = lane & 15, hh = lane >> 4;

  for (int kt = 0; kt < KTILES; ++kt) {
    // A fragments for both 16-row subtiles (2x ds_load_b128 each)
    v16h af0, af1;
    const _Float16* ap0 = As + mrow * ASTR + kt * 32;
    const _Float16* ap1 = As + (16 + mrow) * ASTR + kt * 32;
    #pragma unroll
    for (int v = 0; v < 8; ++v) {
      const int kb = (v < 4) ? (2 * v + 8 * hh) : (16 + 2 * (v - 4) + 8 * hh);
      af0[2 * v]     = ap0[kb];
      af0[2 * v + 1] = ap0[kb + 1];
      af1[2 * v]     = ap1[kb];
      af1[2 * v + 1] = ap1[kb + 1];
    }
    const _Float16* wp =
        W16s + (((size_t)(b * KTILES + kt) * NTILES) + w * 8) * 32 * 16 + lane * 16;
    #pragma unroll
    for (int i = 0; i < 8; ++i) {
      const v16h bfrag = *(const v16h*)(wp + (size_t)i * 32 * 16);
      acc[0][i] = __builtin_amdgcn_wmma_f32_16x16x32_f16(
          false, af0, false, bfrag, (short)0, acc[0][i], false, false);
      acc[1][i] = __builtin_amdgcn_wmma_f32_16x16x32_f16(
          false, af1, false, bfrag, (short)0, acc[1][i], false, false);
    }
  }

  // ---- fold in c[b,:], per-lane row partial sums ----
  const float* __restrict__ cv = c_vec + b * DD;
  float ps[2][8], ps2[2][8];
  #pragma unroll
  for (int mg = 0; mg < 2; ++mg)
    #pragma unroll
    for (int j = 0; j < 8; ++j) { ps[mg][j] = 0.f; ps2[mg][j] = 0.f; }

  #pragma unroll
  for (int i = 0; i < 8; ++i) {
    const int ncol = (w * 8 + i) * 16 + (lane & 15);
    const float cadd = cv[ncol];
    #pragma unroll
    for (int mg = 0; mg < 2; ++mg)
      #pragma unroll
      for (int j = 0; j < 8; ++j) {
        const float x = acc[mg][i][j] + cadd;
        acc[mg][i][j] = x;
        ps[mg][j] += x; ps2[mg][j] += x * x;
      }
  }
  // reduce within 16-lane segment (columns of this wave)
  #pragma unroll
  for (int off = 8; off >= 1; off >>= 1) {
    #pragma unroll
    for (int mg = 0; mg < 2; ++mg)
      #pragma unroll
      for (int j = 0; j < 8; ++j) {
        ps[mg][j]  += __shfl_down(ps[mg][j],  off, 16);
        ps2[mg][j] += __shfl_down(ps2[mg][j], off, 16);
      }
  }
  if ((lane & 15) == 0) {
    #pragma unroll
    for (int mg = 0; mg < 2; ++mg)
      #pragma unroll
      for (int j = 0; j < 8; ++j) {
        const int row = mg * 16 + j + 8 * hh;   // C layout: VGPR j -> M=j (+8 hi half)
        red1[w * 32 + row] = ps[mg][j];
        red2[w * 32 + row] = ps2[mg][j];
      }
  }
  __syncthreads();
  if (t < 32) {
    float s = 0.f, s2 = 0.f;
    for (int w2 = 0; w2 < 12; ++w2) { s += red1[w2 * 32 + t]; s2 += red2[w2 * 32 + t]; }
    tot1[t] = s; tot2[t] = s2;
  }
  __syncthreads();

  // ---- per-row mean/rstd for the 16 rows this lane holds ----
  float lmean[2][8], lrstd[2][8];
  #pragma unroll
  for (int mg = 0; mg < 2; ++mg)
    #pragma unroll
    for (int j = 0; j < 8; ++j) {
      const int row = mg * 16 + j + 8 * hh;
      const float mean = tot1[row] * (1.f / DD);
      const float var  = tot2[row] * (1.f / DD) - mean * mean;
      lmean[mg][j] = mean;
      lrstd[mg][j] = rsqrtf(var + 1e-5f);
    }

  // ---- normalize in registers, store to global ----
  #pragma unroll
  for (int i = 0; i < 8; ++i) {
    const int ncol = (w * 8 + i) * 16 + (lane & 15);
    const float gf = g_fuse[ncol];
    const float bb = b_fuse[ncol];
    #pragma unroll
    for (int mg = 0; mg < 2; ++mg)
      #pragma unroll
      for (int j = 0; j < 8; ++j) {
        const int row = mg * 16 + j + 8 * hh;
        const float val = (acc[mg][i][j] - lmean[mg][j]) * lrstd[mg][j] * gf + bb;
        out[(size_t)(m0 + row) * DD + ncol] = val;
      }
  }
}

// ---------------------------------------------------------------------------
extern "C" void kernel_launch(void* const* d_in, const int* in_sizes, int n_in,
                              void* d_out, int out_size, void* d_ws, size_t ws_size,
                              hipStream_t stream) {
  const float* img    = (const float*)d_in[0];
  const float* pre    = (const float*)d_in[1];
  const int*   cat    = (const int*)  d_in[2];
  const float* Wp     = (const float*)d_in[3];
  const float* bp     = (const float*)d_in[4];
  const float* g_img  = (const float*)d_in[5];
  const float* b_img  = (const float*)d_in[6];
  const float* g_attn = (const float*)d_in[7];
  const float* b_attn = (const float*)d_in[8];
  const float* Wf     = (const float*)d_in[9];
  const float* bf     = (const float*)d_in[10];
  const float* g_fuse = (const float*)d_in[11];
  const float* b_fuse = (const float*)d_in[12];
  float* out = (float*)d_out;

  char* ws = (char*)d_ws;
  float*    p_vec = (float*)ws;                        // 48 KB
  float*    a_vec = (float*)(ws + 64 * 1024);          // 48 KB
  float*    c_vec = (float*)(ws + 128 * 1024);         // 48 KB
  _Float16* W16s  = (_Float16*)(ws + 192 * 1024);      // 37.75 MB

  k_prep<<<PROJ_BLOCKS + WCONV_BLOCKS, 256, 0, stream>>>(img, Wp, bp, Wf, cat, p_vec, W16s);
  k_ln  <<<NB, 256, 0, stream>>>(p_vec, g_img, b_img, g_attn, b_attn, a_vec);
  k_cvec<<<NB * 6, 256, 0, stream>>>(a_vec, Wf, bf, cat, c_vec);
  k_gemm<<<(NB * NN) / 32, 384, 0, stream>>>(pre, W16s, c_vec, g_fuse, b_fuse, out);
}
